// HyperNodeAggregation_4861902979664
// MI455X (gfx1250) — matrix operand order repaired and checked by reading, not compiled, not run on hardware
//
#include <hip/hip_runtime.h>
#include <hip/hip_bf16.h>

typedef __attribute__((ext_vector_type(16))) _Float16 v16h;
typedef __attribute__((ext_vector_type(8)))  float    v8f;
typedef _Float16 f16;

#define WMMA(a, b, c) __builtin_amdgcn_wmma_f32_16x16x32_f16( \
    false, (a), false, (b), (short)0, (c), false, false)

static constexpr int Bn   = 4;
static constexpr int C    = 512;
static constexpr int N    = 4096;   // H*W == NODE_DIM
static constexpr int ND   = 4096;
static constexpr int KAUG = 544;    // 512 + 32 augmented contraction

// ---------- WMMA fragment loaders (ISA 7.12.2 layouts, wave32) ----------
// A matrix 16x32 f16: lanes 0-15 = rows, K split 0..7 / 16..23 ; lanes 16-31 K 8..15 / 24..31
__device__ __forceinline__ v16h load_frag_a(const f16* base, int ld, int r0, int k0) {
    int lane = threadIdx.x & 31;
    int s = (lane >> 4) << 3;                       // 0 or 8
    const f16* p = base + (size_t)(r0 + (lane & 15)) * ld + k0 + s;
    union { v16h v; float4 f[2]; } u;
    u.f[0] = *(const float4*)(p);
    u.f[1] = *(const float4*)(p + 16);
    return u.v;
}
// B matrix 32x16 f16: lane (&15) = column; lanes 0-15 hold K 0..7 / 16..23, lanes 16-31 K 8..15 / 24..31
// memory layout assumed [column][k] contiguous in k (i.e. B stored column-major)
__device__ __forceinline__ v16h load_frag_b(const f16* base, int ld, int c0, int k0) {
    int lane = threadIdx.x & 31;
    int s = (lane >> 4) << 3;                       // 0 or 8
    const f16* p = base + (size_t)(c0 + (lane & 15)) * ld + k0 + s;
    union { v16h v; float4 f[2]; } u;
    u.f[0] = *(const float4*)(p);
    u.f[1] = *(const float4*)(p + 16);
    return u.v;
}

// ---------------- K0a: x -> f16 (row & col major) + row sums ----------------
__global__ __launch_bounds__(256) void cvt_x_kernel(const float* __restrict__ x,
                                                    f16* __restrict__ Xh,
                                                    f16* __restrict__ XhT,
                                                    float* __restrict__ xsum) {
    int bc = blockIdx.x;                  // b*C + c
    int b  = bc >> 9;
    int c  = bc & (C - 1);
    const float* xr = x + (size_t)bc * N;
    f16* Xr = Xh  + (size_t)bc * N;
    f16* XT = XhT + (size_t)b * N * C + c;
    float s = 0.f;
    for (int n = threadIdx.x; n < N; n += 256) {
        float v = xr[n];
        s += v;
        Xr[n] = (f16)v;
        XT[(size_t)n * C] = (f16)v;
    }
    __shared__ float sm[256];
    sm[threadIdx.x] = s;
    __syncthreads();
    for (int off = 128; off; off >>= 1) {
        if (threadIdx.x < off) sm[threadIdx.x] += sm[threadIdx.x + off];
        __syncthreads();
    }
    if (threadIdx.x == 0) xsum[bc] = sm[0];
}

// ---------------- generic f32 -> f16 convert ----------------
__global__ __launch_bounds__(256) void cvt_kernel(const float* __restrict__ src,
                                                  f16* __restrict__ dst, int n) {
    int i = blockIdx.x * 256 + threadIdx.x;
    if (i < n) dst[i] = (f16)src[i];
}

// ---------------- K0b: build augmented Wq rows + w = Wk . xsum ----------------
__global__ __launch_bounds__(256) void build_aug_kernel(const float* __restrict__ Wq,
                                                        const float* __restrict__ bq,
                                                        const float* __restrict__ Wk,
                                                        const float* __restrict__ xsum,
                                                        f16* __restrict__ WqA,
                                                        float* __restrict__ wbuf) {
    int d = blockIdx.x;        // 0..ND-1
    int b = blockIdx.y;        // batch
    const float* xs  = xsum + (size_t)b * C;
    const float* wqr = Wq + (size_t)d * C;
    const float* wkr = Wk + (size_t)d * C;
    f16* dst = WqA + ((size_t)b * ND + d) * KAUG;
    float su = 0.f, sw = 0.f;
    for (int c = threadIdx.x; c < C; c += 256) {
        float xv = xs[c];
        su += wqr[c] * xv;
        sw += wkr[c] * xv;
        dst[c] = (f16)wqr[c];
    }
    __shared__ float s1[256], s2[256];
    s1[threadIdx.x] = su; s2[threadIdx.x] = sw;
    __syncthreads();
    for (int off = 128; off; off >>= 1) {
        if (threadIdx.x < off) { s1[threadIdx.x] += s1[threadIdx.x + off];
                                 s2[threadIdx.x] += s2[threadIdx.x + off]; }
        __syncthreads();
    }
    if (threadIdx.x == 0) {
        dst[512] = (f16)s1[0];
        dst[513] = (f16)bq[d];
        wbuf[(size_t)b * ND + d] = s2[0];
    }
    if (threadIdx.x < 30) dst[514 + threadIdx.x] = (f16)0.f;
}

// ---------------- K2b: fill augmented tail of M^T rows ----------------
__global__ __launch_bounds__(256) void fill_maug_kernel(const float* __restrict__ bk,
                                                        const float* __restrict__ wbuf,
                                                        f16* __restrict__ MTA) {
    int idx = blockIdx.x * 256 + threadIdx.x;     // over Bn*ND
    if (idx >= Bn * ND) return;
    int d = idx & (ND - 1);
    f16* row = MTA + (size_t)idx * KAUG;
    float bkv = bk[d];
    row[512] = (f16)bkv;
    row[513] = (f16)(wbuf[idx] + (float)N * bkv);
    #pragma unroll
    for (int j = 514; j < KAUG; ++j) row[j] = (f16)0.f;
}

// ---------------- K1: G = X X^T (per batch), f16 out ----------------
__global__ __launch_bounds__(32) void xxt_kernel(const f16* __restrict__ Xh,
                                                 f16* __restrict__ Gh) {
    int ct = blockIdx.x, cpt = blockIdx.y, b = blockIdx.z;
    const f16* Xb = Xh + (size_t)b * C * N;
    v8f acc = {};
    for (int k = 0; k < N; k += 32) {
        v16h a  = load_frag_a(Xb, N, ct * 16, k);
        v16h bb = load_frag_b(Xb, N, cpt * 16, k);   // B[k=n][col=c'] = X[c'][n]
        acc = WMMA(a, bb, acc);
    }
    int lane = threadIdx.x & 31, h = lane >> 4, ql = lane & 15;
    f16* Gb = Gh + (size_t)b * C * C;
    #pragma unroll
    for (int i = 0; i < 8; ++i)
        Gb[(size_t)(ct * 16 + i + 8 * h) * C + cpt * 16 + ql] = (f16)acc[i];
}

// ---------------- K2: M = G Wk^T, stored transposed into MTA[:, 0:512] ----------------
__global__ __launch_bounds__(32) void gwk_kernel(const f16* __restrict__ Gh,
                                                 const f16* __restrict__ WkH,
                                                 f16* __restrict__ MTA) {
    int ct = blockIdx.x, dt = blockIdx.y, b = blockIdx.z;
    const f16* Gb = Gh + (size_t)b * C * C;
    v8f acc = {};
    for (int k = 0; k < C; k += 32) {
        v16h a  = load_frag_a(Gb, C, ct * 16, k);
        v16h bb = load_frag_b(WkH, C, dt * 16, k);   // B[k=c'][col=d] = Wk[d][c']
        acc = WMMA(a, bb, acc);
    }
    int lane = threadIdx.x & 31, h = lane >> 4, ql = lane & 15;
    f16* MT = MTA + (size_t)b * ND * KAUG;
    // 8 consecutive halves per lane -> single 16B store
    union { f16 hbuf[8]; float4 f4; } u;
    #pragma unroll
    for (int i = 0; i < 8; ++i) u.hbuf[i] = (f16)acc[i];
    *(float4*)&MT[(size_t)(dt * 16 + ql) * KAUG + ct * 16 + 8 * h] = u.f4;
}

// ---------------- K3: V = Wv X + bv, f16 out ----------------
__global__ __launch_bounds__(32) void vproj_kernel(const f16* __restrict__ WvH,
                                                   const f16* __restrict__ XhT,
                                                   const float* __restrict__ bv,
                                                   f16* __restrict__ Vh) {
    int ct = blockIdx.x, nt = blockIdx.y, b = blockIdx.z;
    const f16* XTb = XhT + (size_t)b * N * C;
    v8f acc = {};
    for (int k = 0; k < C; k += 32) {
        v16h a  = load_frag_a(WvH, C, ct * 16, k);
        v16h bb = load_frag_b(XTb, C, nt * 16, k);   // B[k=c'][col=n] = XhT[n][c']
        acc = WMMA(a, bb, acc);
    }
    int lane = threadIdx.x & 31, h = lane >> 4, ql = lane & 15;
    f16* Vb = Vh + (size_t)b * C * N;
    #pragma unroll
    for (int i = 0; i < 8; ++i) {
        int c = ct * 16 + i + 8 * h;
        Vb[(size_t)c * N + nt * 16 + ql] = (f16)(acc[i] + bv[c]);
    }
}

// ---------------- K4: single-pass online-softmax flash attention ----------------
// grid: (ND/16 q-blocks, Bn); 256 threads = 8 waves. Each workgroup: 16 queries.
// Waves split the 4096 k-columns (32 cols/wave/step, 16 steps); C rows split 64/wave.
__global__ __launch_bounds__(256) void attn_out_kernel(const f16* __restrict__ WqA,
                                                       const f16* __restrict__ MTA,
                                                       const f16* __restrict__ Vh,
                                                       float* __restrict__ out) {
    constexpr int LDP = 264;                      // 256 + 8, bank-conflict-free
    __shared__ f16   Ph[16 * LDP];
    __shared__ float red[8 * 16];
    __shared__ float mrow[16];                    // running row maxima
    __shared__ float scrow[16];                   // exp(m_old - m_new) per row
    __shared__ float srow[16];

    int b = blockIdx.y, q0 = blockIdx.x * 16;
    int wave = threadIdx.x >> 5, lane = threadIdx.x & 31;
    int h = lane >> 4, ql = lane & 15;

    const f16* Ab = WqA + ((size_t)b * ND + q0) * KAUG;
    const f16* Mb = MTA + (size_t)b * ND * KAUG;
    const f16* Vb = Vh  + (size_t)b * C * N;

    v16h afr[17];
    #pragma unroll
    for (int i = 0; i < 17; ++i) afr[i] = load_frag_a(Ab, KAUG, 0, i * 32);

    float sr[8];
    #pragma unroll
    for (int i = 0; i < 8; ++i) sr[i] = 0.f;
    v8f oacc[4];
    #pragma unroll
    for (int t = 0; t < 4; ++t) oacc[t] = (v8f){};

    if (threadIdx.x < 16) mrow[threadIdx.x] = -1e30f;
    __syncthreads();

    for (int step = 0; step < 16; ++step) {
        int kw = step * 256 + wave * 32;
        __builtin_prefetch(Mb + (size_t)(kw + 256) * KAUG, 0, 1);

        // ---- energy tiles for this wave's 32 k-columns (K = 544 augmented)
        v8f s0 = {}, s1 = {};
        #pragma unroll
        for (int i = 0; i < 17; ++i) {
            v16h bfr0 = load_frag_b(Mb, KAUG, kw, i * 32);
            s0 = WMMA(afr[i], bfr0, s0);
            v16h bfr1 = load_frag_b(Mb, KAUG, kw + 16, i * 32);
            s1 = WMMA(afr[i], bfr1, s1);
        }

        // ---- per-row max over this wave's 32 columns
        float rm[8];
        #pragma unroll
        for (int i = 0; i < 8; ++i) {
            rm[i] = fmaxf(s0[i], s1[i]);
            for (int off = 1; off < 16; off <<= 1)
                rm[i] = fmaxf(rm[i], __shfl_xor(rm[i], off, 16));
        }
        if (ql == 0) {
            #pragma unroll
            for (int i = 0; i < 8; ++i) red[wave * 16 + i + 8 * h] = rm[i];
        }
        __syncthreads();
        if (threadIdx.x < 16) {
            float m_old = mrow[threadIdx.x];
            float m = m_old;
            for (int w = 0; w < 8; ++w) m = fmaxf(m, red[w * 16 + threadIdx.x]);
            mrow[threadIdx.x]  = m;
            scrow[threadIdx.x] = __expf(m_old - m);   // 0 on first step
        }
        __syncthreads();

        // ---- rescale out accumulators (out-tile column = q = ql)
        float scq = scrow[ql];
        #pragma unroll
        for (int t2 = 0; t2 < 4; ++t2)
            #pragma unroll
            for (int i = 0; i < 8; ++i) oacc[t2][i] *= scq;

        // ---- P = exp(S - m_new), rescale running sums, stage P in LDS
        #pragma unroll
        for (int i = 0; i < 8; ++i) {
            int row = i + 8 * h;
            float mn = mrow[row];
            float sc = scrow[row];
            float p0 = __expf(s0[i] - mn);
            float p1 = __expf(s1[i] - mn);
            sr[i] = sr[i] * sc + p0 + p1;
            Ph[row * LDP + kw - step * 256 + ql]      = (f16)p0;
            Ph[row * LDP + kw - step * 256 + 16 + ql] = (f16)p1;
        }
        __syncthreads();   // all waves' P tiles visible

        // ---- out accumulation: oacc += V[:,k-chunk] * P^T
        #pragma unroll
        for (int kc = 0; kc < 8; ++kc) {
            v16h bp = load_frag_b(Ph, LDP, 0, kc * 32);   // B[k][q] = P[q][k]
            #pragma unroll
            for (int t2 = 0; t2 < 4; ++t2) {
                v16h av = load_frag_a(Vb, N, wave * 64 + t2 * 16, step * 256 + kc * 32);
                oacc[t2] = WMMA(av, bp, oacc[t2]);
            }
        }
        __syncthreads();   // done reading Ph before next step overwrites
    }

    // ---- merge row sums across lanes and waves, scale & store output
    #pragma unroll
    for (int i = 0; i < 8; ++i)
        for (int off = 1; off < 16; off <<= 1)
            sr[i] += __shfl_xor(sr[i], off, 16);
    if (ql == 0) {
        #pragma unroll
        for (int i = 0; i < 8; ++i) red[wave * 16 + i + 8 * h] = sr[i];
    }
    __syncthreads();
    if (threadIdx.x < 16) {
        float s = 0.f;
        for (int w = 0; w < 8; ++w) s += red[w * 16 + threadIdx.x];
        srow[threadIdx.x] = s;
    }
    __syncthreads();
    float inv = 1.0f / srow[ql];
    float* ob = out + (size_t)b * C * N;
    #pragma unroll
    for (int t2 = 0; t2 < 4; ++t2)
        #pragma unroll
        for (int i = 0; i < 8; ++i) {
            int c = wave * 64 + t2 * 16 + i + 8 * h;
            ob[(size_t)c * N + q0 + ql] = oacc[t2][i] * inv;
        }
}

// ---------------- host launcher ----------------
extern "C" void kernel_launch(void* const* d_in, const int* in_sizes, int n_in,
                              void* d_out, int out_size, void* d_ws, size_t ws_size,
                              hipStream_t stream) {
    const float* x  = (const float*)d_in[0];
    const float* Wq = (const float*)d_in[1];
    const float* bq = (const float*)d_in[2];
    const float* Wk = (const float*)d_in[3];
    const float* bk = (const float*)d_in[4];
    const float* Wv = (const float*)d_in[5];
    const float* bv = (const float*)d_in[6];
    float* out = (float*)d_out;

    char* ws = (char*)d_ws;
    size_t off = 0;
    auto alloc = [&](size_t bytes) { size_t o = off; off = (off + bytes + 255) & ~(size_t)255; return o; };

    f16*   Xh   = (f16*)  (ws + alloc((size_t)Bn * C * N * 2));
    f16*   XhT  = (f16*)  (ws + alloc((size_t)Bn * N * C * 2));
    f16*   WkH  = (f16*)  (ws + alloc((size_t)ND * C * 2));
    f16*   WvH  = (f16*)  (ws + alloc((size_t)C * C * 2));
    float* xsum = (float*)(ws + alloc((size_t)Bn * C * 4));
    f16*   Gh   = (f16*)  (ws + alloc((size_t)Bn * C * C * 2));
    f16*   WqA  = (f16*)  (ws + alloc((size_t)Bn * ND * KAUG * 2));
    f16*   MTA  = (f16*)  (ws + alloc((size_t)Bn * ND * KAUG * 2));
    float* wbuf = (float*)(ws + alloc((size_t)Bn * ND * 4));
    f16*   Vh   = (f16*)  (ws + alloc((size_t)Bn * C * N * 2));
    (void)ws_size; (void)in_sizes; (void)n_in; (void)out_size;

    // Stage 0: conversions + row sums
    cvt_x_kernel<<<Bn * C, 256, 0, stream>>>(x, Xh, XhT, xsum);
    cvt_kernel<<<(ND * C + 255) / 256, 256, 0, stream>>>(Wk, WkH, ND * C);
    cvt_kernel<<<(C * C + 255) / 256, 256, 0, stream>>>(Wv, WvH, C * C);
    // Stage 0b: augmented Wq rows (u = Wq.xsum folded in) + w = Wk.xsum
    build_aug_kernel<<<dim3(ND, Bn), 256, 0, stream>>>(Wq, bq, Wk, xsum, WqA, wbuf);
    // Stage 1: G = X X^T
    xxt_kernel<<<dim3(C / 16, C / 16, Bn), 32, 0, stream>>>(Xh, Gh);
    // Stage 2: M^T = (G Wk^T)^T into augmented buffer; then fill bias tail
    gwk_kernel<<<dim3(C / 16, ND / 16, Bn), 32, 0, stream>>>(Gh, WkH, MTA);
    fill_maug_kernel<<<(Bn * ND + 255) / 256, 256, 0, stream>>>(bk, wbuf, MTA);
    // Stage 3: V = Wv X + bv
    vproj_kernel<<<dim3(C / 16, N / 16, Bn), 32, 0, stream>>>(WvH, XhT, bv, Vh);
    // Stage 4: single-pass online-softmax flash attention
    attn_out_kernel<<<dim3(ND / 16, Bn), 256, 0, stream>>>(WqA, MTA, Vh, out);
}
